// QLoRALinear_18193481466076
// MI455X (gfx1250) — matrix operand order repaired and checked by reading, not compiled, not run on hardware
//
#include <hip/hip_runtime.h>

// QLoRA linear for MI455X (gfx1250, wave32).
// M=512 (B*S), K=4096 (IN), N=11008 (OUT). NF4 dequant fused into a
// bf16 WMMA GEMM (v_wmma_f32_16x16x32_bf16); LoRA rank-16 in the epilogue.
//
// Roofline: 46 GFLOP vs 213 MB mandatory traffic (180 MB int32 NF4 indices
// dominate) -> ~9 us at 23.3 TB/s. Index stream is read 16 times, once per 32-row M-block (each
// wave owns a 16-col N-tile within its block). Each dequantized B fragment is reused
// by TWO M-tiles (independent accumulators -> no WMMA RAW serialization),
// halving per-WMMA dequant DS/VALU cost. Dequant itself: 256-entry packed
// bf16 pair-LUT in LDS + v_pk_mul_bf16 with broadcast scale.

#define IN_DIM  4096
#define OUT_DIM 11008
#define M_DIM   512
#define RANK    16

typedef __attribute__((ext_vector_type(16))) __bf16 bf16x16;
typedef __attribute__((ext_vector_type(2)))  __bf16 bf16x2;
typedef __attribute__((ext_vector_type(8)))  float  f32x8;

__constant__ float NF4_LUT[16] = {
  -1.0f, -0.6961928009986877f, -0.5250730514526367f, -0.39491748809814453f,
  -0.28444138169288635f, -0.18477343022823334f, -0.09105003625154495f, 0.0f,
   0.07958029955625534f, 0.16093020141124725f, 0.24611230194568634f,
   0.33791524171829224f, 0.44070982933044434f, 0.5626170039176941f,
   0.7229568362236023f, 1.0f };

// packed f32x2 -> bf16x2 (lowers to v_cvt_pk_bf16_f32)
__device__ __forceinline__ unsigned cvt_pk_bf16(float lo, float hi) {
  union { unsigned u; bf16x2 h; } r;
  r.h.x = (__bf16)lo;
  r.h.y = (__bf16)hi;
  return r.u;
}

// packed bf16 pair * broadcast bf16 scale (v_pk_mul_bf16)
__device__ __forceinline__ unsigned pk_scale(unsigned pair, bf16x2 ss) {
  union { unsigned u; bf16x2 h; } a;
  a.u = pair;
  a.h = a.h * ss;
  return a.u;
}

// ---------------------------------------------------------------------------
// Kernel 1: T[m, r] = sum_k x[m,k] * lora_A[k,r]   (512 x 16, tiny)
// ---------------------------------------------------------------------------
__global__ __launch_bounds__(256)
void lora_xA_kernel(const float* __restrict__ x,
                    const float* __restrict__ loraA,
                    float* __restrict__ T) {
  const int m = blockIdx.x;
  const int tid = threadIdx.x;
  float acc[RANK];
#pragma unroll
  for (int r = 0; r < RANK; ++r) acc[r] = 0.f;

  const size_t xrow = (size_t)m * IN_DIM;
  for (int k = tid; k < IN_DIM; k += 256) {
    const float xv = x[xrow + k];
    const float* Ar = loraA + (size_t)k * RANK;
#pragma unroll
    for (int r = 0; r < RANK; ++r) acc[r] += xv * Ar[r];
  }

  __shared__ float red[256];
#pragma unroll 1
  for (int r = 0; r < RANK; ++r) {
    red[tid] = acc[r];
    __syncthreads();
    for (int s = 128; s > 0; s >>= 1) {
      if (tid < s) red[tid] += red[tid + s];
      __syncthreads();
    }
    if (tid == 0) T[m * RANK + r] = red[0];
    __syncthreads();
  }
}

// ---------------------------------------------------------------------------
// Kernel 2: fused NF4-dequant bf16 WMMA GEMM + bias + LoRA epilogue.
// Block = 256 threads = 8 waves; block owns 32 M-rows (two 16-row tiles),
// wave w owns N-tile (blockIdx.x*8 + w)*16 and reuses each dequantized B
// fragment for both M-tiles. x tile (32x64) staged f32->bf16 in
// double-buffered LDS: two barriers + one scale load per FOUR wmma ops.
// ---------------------------------------------------------------------------
__global__ __launch_bounds__(256)
void qlora_wmma_kernel(const float* __restrict__ x,
                       const int*   __restrict__ widx,
                       const float* __restrict__ wscale,
                       const float* __restrict__ loraB,
                       const float* __restrict__ bias,
                       const float* __restrict__ T,
                       float* __restrict__ out) {
  __shared__ __align__(16) unsigned xs32[2][1024];  // 2 x 4KB: 32 rows x 64 K bf16
  __shared__ unsigned lut2[256];                    // packed bf16 (NF4[lo],NF4[hi])

  const int tid  = threadIdx.x;
  const int wave = tid >> 5;
  const int lane = tid & 31;
  const int m0   = blockIdx.y * 32;
  const int n0   = (blockIdx.x * 8 + wave) * 16;

  // 256-entry pair LUT: addr = lo | (hi<<4), value = bf16x2(NF4[lo], NF4[hi])
  lut2[tid] = cvt_pk_bf16(NF4_LUT[tid & 15], NF4_LUT[tid >> 4]);

  const int nl = lane & 15;            // N within tile (B/C lane = column)
  const int hg = lane >> 4;            // lane half-group (K split)
  const int o  = n0 + nl;              // output column this lane dequantizes
  const size_t orow = (size_t)o * IN_DIM;

  // staging: 1024 float2 pairs (32 rows x 32 pairs), 4 per thread
  const int cp   = tid & 31;           // pair-column 0..31
  const int rbase = tid >> 5;          // rows rbase, rbase+8, rbase+16, rbase+24
  const float2* xf[4];
#pragma unroll
  for (int j = 0; j < 4; ++j)
    xf[j] = reinterpret_cast<const float2*>(x + (size_t)(m0 + rbase + 8 * j) * IN_DIM);

  f32x8 c0 = {0.f, 0.f, 0.f, 0.f, 0.f, 0.f, 0.f, 0.f};
  f32x8 c1 = {0.f, 0.f, 0.f, 0.f, 0.f, 0.f, 0.f, 0.f};

  const char* xsbase = reinterpret_cast<const char*>(&xs32[0][0]);
  // A-frag byte offset inside one 4KB buffer (row stride = 128 B):
  //   tile0: row = lane&15, tile1: +16 rows (=+2048 B); koff = hg*16 B
  const int abyte = ((lane & 15) << 7) + (hg << 4);

  for (int k0 = 0; k0 < IN_DIM; k0 += 64) {
    const int buf = (k0 >> 6) & 1;

    // ---- cooperative stage: 32x64 x-tile, f32 -> packed bf16 in LDS ----
#pragma unroll
    for (int j = 0; j < 4; ++j) {
      const float2 f = xf[j][(k0 >> 1) + cp];
      xs32[buf][((rbase + 8 * j) << 5) + cp] = cvt_pk_bf16(f.x, f.y);
    }
    __syncthreads();

    // one NF4 scale block covers all 64 K of this step
    const float s = wscale[(o << 6) + (k0 >> 6)];
    bf16x2 ss;
    ss.x = (__bf16)s;
    ss.y = ss.x;

    const int4* ip = reinterpret_cast<const int4*>(widx + orow + k0 + (hg << 3));
    if (k0 + 64 < IN_DIM)
      __builtin_prefetch(widx + orow + k0 + 64 + (hg << 4), 0, 0);

#pragma unroll
    for (int sub = 0; sub < 2; ++sub) {
      // ---- A fragments for both M-tiles from LDS ----
      union { bf16x16 v; uint4 q[2]; } a0, a1;
      const char* abp = xsbase + (size_t)buf * 4096 + abyte + (sub << 6);
      a0.q[0] = *reinterpret_cast<const uint4*>(abp);            // K koff..+7
      a0.q[1] = *reinterpret_cast<const uint4*>(abp + 32);       // K koff+16..+23
      a1.q[0] = *reinterpret_cast<const uint4*>(abp + 2048);
      a1.q[1] = *reinterpret_cast<const uint4*>(abp + 2048 + 32);

      // ---- B fragment: dequant 16 NF4 weights of column o via pair LUT ----
      const int4 i0 = ip[sub * 8 + 0];
      const int4 i1 = ip[sub * 8 + 1];
      const int4 i2 = ip[sub * 8 + 4];
      const int4 i3 = ip[sub * 8 + 5];
      union { bf16x16 v; unsigned w[8]; } b;
      b.w[0] = pk_scale(lut2[i0.x | (i0.y << 4)], ss);
      b.w[1] = pk_scale(lut2[i0.z | (i0.w << 4)], ss);
      b.w[2] = pk_scale(lut2[i1.x | (i1.y << 4)], ss);
      b.w[3] = pk_scale(lut2[i1.z | (i1.w << 4)], ss);
      b.w[4] = pk_scale(lut2[i2.x | (i2.y << 4)], ss);
      b.w[5] = pk_scale(lut2[i2.z | (i2.w << 4)], ss);
      b.w[6] = pk_scale(lut2[i3.x | (i3.y << 4)], ss);
      b.w[7] = pk_scale(lut2[i3.z | (i3.w << 4)], ss);

      // Two independent accumulator chains share one dequantized B.
      c0 = __builtin_amdgcn_wmma_f32_16x16x32_bf16(false, a0.v, false, b.v,
                                                   (short)0, c0, false, false);
      c1 = __builtin_amdgcn_wmma_f32_16x16x32_bf16(false, a1.v, false, b.v,
                                                   (short)0, c1, false, false);
    }
    __syncthreads();  // all A-frag reads done before next-stage overwrite
  }

  // ---- epilogue: + bias + 2.0 * (T @ loraB), fused into C-fragment stores ----
  const float bo = bias[o];
  float lb[RANK];
#pragma unroll
  for (int r = 0; r < RANK; ++r) lb[r] = loraB[(size_t)r * OUT_DIM + o];

#pragma unroll
  for (int t = 0; t < 2; ++t) {
    const f32x8& c = t ? c1 : c0;
#pragma unroll
    for (int v = 0; v < 8; ++v) {
      const int m = m0 + 16 * t + v + (hg << 3);  // C layout: M = vgpr + 8*(lane/16)
      const float* Tm = T + m * RANK;
      float acc = 0.f;
#pragma unroll
      for (int r = 0; r < RANK; ++r) acc += Tm[r] * lb[r];
      out[(size_t)m * OUT_DIM + o] = c[v] + bo + 2.0f * acc;
    }
  }
}

// ---------------------------------------------------------------------------
extern "C" void kernel_launch(void* const* d_in, const int* in_sizes, int n_in,
                              void* d_out, int out_size, void* d_ws, size_t ws_size,
                              hipStream_t stream) {
  (void)in_sizes; (void)n_in; (void)out_size; (void)ws_size;
  const float* x      = (const float*)d_in[0];
  const int*   widx   = (const int*)d_in[1];
  const float* wscale = (const float*)d_in[2];
  const float* loraA  = (const float*)d_in[3];
  const float* loraB  = (const float*)d_in[4];
  const float* bias   = (const float*)d_in[5];
  float* out = (float*)d_out;
  float* T   = (float*)d_ws;   // 512*16 floats = 32 KB scratch

  lora_xA_kernel<<<dim3(M_DIM), dim3(256), 0, stream>>>(x, loraA, T);

  qlora_wmma_kernel<<<dim3(OUT_DIM / 128, M_DIM / 32), dim3(256), 0, stream>>>(
      x, widx, wscale, loraB, bias, T, out);
}
